// GraphAttentionLayer_41137196761536
// MI455X (gfx1250) — matrix operand order repaired and checked by reading, not compiled, not run on hardware
//
#include <hip/hip_runtime.h>
#include <hip/hip_bf16.h>

typedef __attribute__((ext_vector_type(16))) __bf16 v16bf;
typedef __attribute__((ext_vector_type(8)))  float  v8f;

#define N_NODES 4096
#define IN_F    256
#define H_HEADS 4
#define F_OUT   64
#define SLOPE   0.2f

union BfVec { v16bf v; uint4 q[2]; };

static __device__ inline void put4(v16bf& v, int base, float4 f) {
    v[base + 0] = (__bf16)f.x;
    v[base + 1] = (__bf16)f.y;
    v[base + 2] = (__bf16)f.z;
    v[base + 3] = (__bf16)f.w;
}

// --- CDNA5 async memory->LDS helpers (ASYNCcnt-tracked, no VGPR round-trip) --
// LDS operand is the wave-relative byte offset = low 32 bits of the generic
// shared-memory pointer (ISA: LDS aperture uses addr[31:0]).
static __device__ inline unsigned int lds_off(const void* p) {
    return (unsigned int)(unsigned long long)(uintptr_t)p;
}
static __device__ inline void async_ld_b128(unsigned int lds, const void* gptr) {
    asm volatile("global_load_async_to_lds_b128 %0, %1, off"
                 :: "v"(lds), "v"((unsigned long long)(uintptr_t)gptr)
                 : "memory");
}
static __device__ inline void async_ld_b32(unsigned int lds, const void* gptr) {
    asm volatile("global_load_async_to_lds_b32 %0, %1, off"
                 :: "v"(lds), "v"((unsigned long long)(uintptr_t)gptr)
                 : "memory");
}
static __device__ inline void wait_async0() {
    asm volatile("s_wait_asynccnt 0x0" ::: "memory");
}

// ---------------------------------------------------------------------------
// Kernel 1: projection g = h @ W^T + b  (bf16 WMMA, f32 accumulate)
//   - one workgroup = 16 rows of g; wave w = head w (64 output cols)
//   - writes gT (transposed bf16 [256][4096]) for the aggregation B-tiles
//   - computes src[i,h] = <g_i, aw[:64]>, dst[i,h] = <g_i, aw[64:]> exactly
// ---------------------------------------------------------------------------
__global__ __launch_bounds__(128)
void gat_proj_kernel(const float* __restrict__ h,
                     const float* __restrict__ W,
                     const float* __restrict__ bias,
                     const float* __restrict__ attn_w,
                     unsigned short* __restrict__ gT,
                     float* __restrict__ srcv,
                     float* __restrict__ dstv)
{
    const int i0   = blockIdx.x * 16;
    const int wave = threadIdx.x >> 5;   // head
    const int lane = threadIdx.x & 31;
    const int m    = lane & 15;          // A row / B col within tile
    const int kh   = lane >> 4;          // lane half

    v8f acc[4];
    #pragma unroll
    for (int t = 0; t < 4; ++t)
        #pragma unroll
        for (int r = 0; r < 8; ++r) acc[t][r] = 0.f;

    #pragma unroll
    for (int kk = 0; kk < IN_F / 32; ++kk) {
        const int k0 = kk * 32;

        // A tile: h[i0+m][k0 + {16kh..16kh+15}]
        v16bf A;
        const float4* h4 = (const float4*)(h + (size_t)(i0 + m) * IN_F + k0);
        put4(A, 0,  h4[4 * kh]);
        put4(A, 4,  h4[4 * kh + 1]);
        put4(A, 8,  h4[4 * kh + 2]);
        put4(A, 12, h4[4 * kh + 3]);

        #pragma unroll
        for (int nt = 0; nt < 4; ++nt) {
            // B[k][n] = W[n][k]; lane holds n = wave*64+nt*16+m, K = comp + 16*kh
            const int n = wave * 64 + nt * 16 + m;
            const float4* w4 = (const float4*)(W + (size_t)n * IN_F + k0 + 16 * kh);
            v16bf B;
            put4(B, 0,  w4[0]);
            put4(B, 4,  w4[1]);
            put4(B, 8,  w4[2]);
            put4(B, 12, w4[3]);
            acc[nt] = __builtin_amdgcn_wmma_f32_16x16x32_bf16(
                false, A, false, B, (short)0, acc[nt], false, false);
        }
    }

    // epilogue: bias, gT store (bf16, transposed), src/dst partials
    float awS[4], awD[4];
    #pragma unroll
    for (int nt = 0; nt < 4; ++nt) {
        const int f = nt * 16 + m;            // feature index within head
        awS[nt] = attn_w[f];
        awD[nt] = attn_w[F_OUT + f];
    }
    float ps[8], pd[8];
    #pragma unroll
    for (int r = 0; r < 8; ++r) { ps[r] = 0.f; pd[r] = 0.f; }

    #pragma unroll
    for (int nt = 0; nt < 4; ++nt) {
        const int n  = wave * 64 + nt * 16 + m;   // global g column
        const float bv = bias[n];
        #pragma unroll
        for (int r = 0; r < 8; ++r) {
            const float val = acc[nt][r] + bv;
            const int row = r + 8 * kh;           // C layout: lanes16-31 -> M+8
            const __bf16 bf = (__bf16)val;
            gT[(size_t)n * N_NODES + i0 + row] = __builtin_bit_cast(unsigned short, bf);
            ps[r] += val * awS[nt];
            pd[r] += val * awD[nt];
        }
    }

    // reduce over 16 lanes within each half (masks 1,2,4,8 stay in-half)
    #pragma unroll
    for (int r = 0; r < 8; ++r) {
        float s = ps[r], d = pd[r];
        #pragma unroll
        for (int mask = 1; mask <= 8; mask <<= 1) {
            s += __shfl_xor(s, mask, 32);
            d += __shfl_xor(d, mask, 32);
        }
        if (m == 0) {
            const int row = i0 + r + 8 * kh;
            srcv[row * H_HEADS + wave] = s;
            dstv[row * H_HEADS + wave] = d;
        }
    }
}

// ---------------------------------------------------------------------------
// Kernel 2: per-head global max of dst  (safe softmax bound)
// ---------------------------------------------------------------------------
__global__ __launch_bounds__(128)
void gat_gmax_kernel(const float* __restrict__ dstv, float* __restrict__ gmax)
{
    __shared__ float red[128];
    const int t  = threadIdx.x;
    const int hh = t & 3;
    float mx = -1e30f;
    for (int j = t >> 2; j < N_NODES; j += 32)
        mx = fmaxf(mx, dstv[j * H_HEADS + hh]);
    red[t] = mx;
    __syncthreads();
    for (int s = 64; s >= 4; s >>= 1) {
        if (t < s) red[t] = fmaxf(red[t], red[t + s]);
        __syncthreads();
    }
    if (t < 4) gmax[t] = red[t];
}

// ---------------------------------------------------------------------------
// Kernel 3: fused masked-softmax attention aggregation (flash-style)
//   out[i,f] = (1/H) * sum_h  (1/Z_ih) * sum_j p_ijh * g[j, h*64+f]
//   p_ijh = adj[i][j] ? exp(lrelu(src_i+dst_j+b) - m_ih) : 0
//   m_ih  = lrelu(src_i + gmax_h + b)  >= all e_ij  (lrelu monotone)
// One workgroup = 16 rows; wave = head; K-loop over 4096 cols in steps of 32.
// Tiles staged with double-buffered GLOBAL_LOAD_ASYNC_TO_LDS (ASYNCcnt),
// overlapping the next tile's fill with the current tile's exp+WMMA work.
// ---------------------------------------------------------------------------
__global__ __launch_bounds__(128)
void gat_attn_agg_kernel(const int* __restrict__ adj,
                         const unsigned short* __restrict__ gT,
                         const float* __restrict__ srcv,
                         const float* __restrict__ dstv,
                         const float* __restrict__ gmaxp,
                         const float* __restrict__ attn_b,
                         float* __restrict__ out)
{
    __shared__ __align__(16) int            adjT[2][16][32];        // 4 KB
    __shared__ __align__(16) float          dstT[2][H_HEADS][32];   // 1 KB
    __shared__ __align__(16) unsigned short gTl[2][256][32];        // 32 KB bf16
    __shared__ float outacc[16][F_OUT];                             // 4 KB
    __shared__ float zbuf[H_HEADS][16];                             // 256 B

    const int i0   = blockIdx.x * 16;
    const int tid  = threadIdx.x;
    const int head = tid >> 5;
    const int lane = tid & 31;
    const int m    = lane & 15;
    const int kh   = lane >> 4;

    const float ab = attn_b[0];
    const float sm = srcv[(size_t)(i0 + m) * H_HEADS + head] + ab;
    const float xm = sm + gmaxp[head];
    const float mi = (xm >= 0.f) ? xm : SLOPE * xm;   // safe row max

    for (int s = tid; s < 16 * F_OUT; s += 128) ((float*)outacc)[s] = 0.f;

    // issue one iteration's tile fills into LDS buffer `buf` (async, no VGPRs)
    const int srow = tid >> 3, sc4 = tid & 7;   // adj tile split: 128 x b128
    const int shh  = tid >> 5, sk  = tid & 31;  // dst tile split: 128 x b32
    auto stage = [&](int buf, int j0) {
        async_ld_b128(lds_off(&adjT[buf][srow][sc4 * 4]),
                      adj + (size_t)(i0 + srow) * N_NODES + j0 + sc4 * 4);
        async_ld_b32(lds_off(&dstT[buf][shh][sk]),
                     dstv + (size_t)(j0 + sk) * H_HEADS + shh);
        #pragma unroll
        for (int s = 0; s < 8; ++s) {           // gT tile: 1024 x b128
            const int id = tid + 128 * s;
            const int c = id >> 2, part = id & 3;
            async_ld_b128(lds_off(&gTl[buf][c][part * 8]),
                          gT + (size_t)c * N_NODES + j0 + part * 8);
        }
    };

    v8f acc[4];
    #pragma unroll
    for (int t = 0; t < 4; ++t)
        #pragma unroll
        for (int r = 0; r < 8; ++r) acc[t][r] = 0.f;
    float zsum = 0.f;

    constexpr int NB = N_NODES / 32;
    stage(0, 0);                                 // prologue fill

    for (int jb = 0; jb < NB; ++jb) {
        const int cur = jb & 1;
        wait_async0();        // this wave's fills for buf[cur] landed in LDS
        __syncthreads();      // all waves' fills landed; prev reads of buf[cur^1] done
        if (jb + 1 < NB) stage(cur ^ 1, (jb + 1) * 32);   // overlap next fill

        // A tile: probabilities in exact ISA A-layout --------------------
        // lane covers row m, K-runs {8kh..8kh+7} and {16+8kh..16+8kh+7}
        v16bf A;
        const int4*   a4 = (const int4*)&adjT[cur][m][0];
        const float4* d4 = (const float4*)&dstT[cur][head][0];
        auto quad = [&](int4 av, float4 dv, int base) {
            float x0 = sm + dv.x, x1 = sm + dv.y, x2 = sm + dv.z, x3 = sm + dv.w;
            x0 = (x0 >= 0.f) ? x0 : SLOPE * x0;
            x1 = (x1 >= 0.f) ? x1 : SLOPE * x1;
            x2 = (x2 >= 0.f) ? x2 : SLOPE * x2;
            x3 = (x3 >= 0.f) ? x3 : SLOPE * x3;
            const float p0 = av.x ? __expf(x0 - mi) : 0.f;
            const float p1 = av.y ? __expf(x1 - mi) : 0.f;
            const float p2 = av.z ? __expf(x2 - mi) : 0.f;
            const float p3 = av.w ? __expf(x3 - mi) : 0.f;
            zsum += p0 + p1 + p2 + p3;
            A[base + 0] = (__bf16)p0;
            A[base + 1] = (__bf16)p1;
            A[base + 2] = (__bf16)p2;
            A[base + 3] = (__bf16)p3;
        };
        quad(a4[2 * kh],     d4[2 * kh],     0);
        quad(a4[2 * kh + 1], d4[2 * kh + 1], 4);
        quad(a4[4 + 2 * kh], d4[4 + 2 * kh], 8);
        quad(a4[5 + 2 * kh], d4[5 + 2 * kh], 12);

        // 4 WMMAs: C[16x64] += P[16x32] * G[32x64] ------------------------
        #pragma unroll
        for (int nt = 0; nt < 4; ++nt) {
            const int coln = head * F_OUT + nt * 16 + m;  // B lane col
            BfVec B;
            const uint4* g4 = (const uint4*)&gTl[cur][coln][0];
            B.q[0] = g4[kh];           // K = 8*kh .. 8*kh+7
            B.q[1] = g4[2 + kh];       // K = 16+8*kh .. 16+8*kh+7
            acc[nt] = __builtin_amdgcn_wmma_f32_16x16x32_bf16(
                false, A, false, B.v, (short)0, acc[nt], false, false);
        }
    }

    // row denominators: lane m and lane m+16 hold complementary K-halves
    const float zfull = zsum + __shfl_xor(zsum, 16, 32);
    if (lane < 16) zbuf[head][lane] = zfull;
    __syncthreads();

    // scale by 1/(H*Z) and merge heads via LDS float atomics
    #pragma unroll
    for (int r = 0; r < 8; ++r) {
        const int row = r + 8 * kh;                 // C layout row
        const float wsc = 0.25f / zbuf[head][row];
        #pragma unroll
        for (int nt = 0; nt < 4; ++nt)
            atomicAdd(&outacc[row][nt * 16 + m], acc[nt][r] * wsc);
    }
    __syncthreads();

    // store 16x64 f32 tile, vectorized
    for (int s = tid; s < 16 * F_OUT / 4; s += 128) {
        const int row = s >> 4, c4 = s & 15;
        *(float4*)(out + (size_t)(i0 + row) * F_OUT + c4 * 4) =
            ((const float4*)outacc)[s];
    }
}

// ---------------------------------------------------------------------------
extern "C" void kernel_launch(void* const* d_in, const int* in_sizes, int n_in,
                              void* d_out, int out_size, void* d_ws, size_t ws_size,
                              hipStream_t stream) {
    (void)in_sizes; (void)n_in; (void)out_size; (void)ws_size;

    const float* h      = (const float*)d_in[0];   // [4096,256]
    const int*   adj    = (const int*)  d_in[1];   // [4096,4096]
    const float* W      = (const float*)d_in[2];   // [256,256]
    const float* b      = (const float*)d_in[3];   // [256]
    const float* attn_w = (const float*)d_in[4];   // [128]
    const float* attn_b = (const float*)d_in[5];   // [1]
    float*       out    = (float*)d_out;           // [4096,64]

    char* wsb = (char*)d_ws;
    unsigned short* gT  = (unsigned short*)wsb;                      // 2 MB bf16 [256][4096]
    float* srcv = (float*)(wsb + (size_t)IN_F * N_NODES * 2);        // 64 KB [4096][4]
    float* dstv = srcv + (size_t)N_NODES * H_HEADS;                  // 64 KB [4096][4]
    float* gmax = dstv + (size_t)N_NODES * H_HEADS;                  // 16 B  [4]

    gat_proj_kernel<<<N_NODES / 16, 128, 0, stream>>>(h, W, b, attn_w, gT, srcv, dstv);
    gat_gmax_kernel<<<1, 128, 0, stream>>>(dstv, gmax);
    gat_attn_agg_kernel<<<N_NODES / 16, 128, 0, stream>>>(adj, gT, srcv, dstv, gmax,
                                                          attn_b, out);
}
